// DeepseekOCRLayer_59167469470472
// MI455X (gfx1250) — matrix operand order repaired and checked by reading, not compiled, not run on hardware
//
#include <hip/hip_runtime.h>

#define T      1024
#define HDIM   1280
#define NHEAD  10
#define HD     128
#define NEXP   64
#define EI     896
#define SHF    1792
#define GU_N   (EI * NEXP)   // 57344
#define DPE_STRIDE (HDIM * NEXP) // 81920

typedef __attribute__((ext_vector_type(16))) __bf16 v16bf;
typedef __attribute__((ext_vector_type(8)))  float  v8f;

// ---------- CDNA5 TDM availability (device pass only; host pass falls back) ----------
#if defined(__has_builtin)
#if __has_builtin(__builtin_amdgcn_tensor_load_to_lds) && __has_builtin(__builtin_amdgcn_s_wait_tensorcnt)
#define USE_TDM 1
#endif
#endif
#ifndef USE_TDM
#define USE_TDM 0
#endif

// ---------- bf16 helpers (round-to-nearest-even) ----------
__device__ __forceinline__ unsigned short f2bf(float f) {
    unsigned u = __float_as_uint(f);
    return (unsigned short)((u + 0x7fffu + ((u >> 16) & 1u)) >> 16);
}
__device__ __forceinline__ unsigned pack2bf(float lo, float hi) {
    return (unsigned)f2bf(lo) | ((unsigned)f2bf(hi) << 16);
}
__device__ __forceinline__ float bf2f(unsigned short s) {
    return __uint_as_float(((unsigned)s) << 16);
}

union U16 { v16bf v; unsigned int u[8]; };

#if USE_TDM
typedef unsigned int u32x4 __attribute__((ext_vector_type(4)));
typedef int i32x8v __attribute__((ext_vector_type(8)));
typedef int i32x4v __attribute__((ext_vector_type(4)));

// Issue a TDM 2-D tile load: 128 rows x 32 bf16 elements, row stride K elements,
// packed contiguously into LDS at lds_off. D# fields per CDNA5 ISA ch.8.
__device__ __forceinline__ void tdm_load_A(const unsigned short* gsrc,
                                           unsigned lds_off, int K_) {
    unsigned long long ga = (unsigned long long)(uintptr_t)gsrc;
    u32x4 g0;
    g0[0] = 1u;                                    // count=1 (valid user D#)
    g0[1] = lds_off;                               // lds_addr (bytes)
    g0[2] = (unsigned)(ga & 0xffffffffu);          // global_addr lo
    g0[3] = (unsigned)((ga >> 32) & 0x1ffffffu) | (2u << 30); // addr hi | type=2
    i32x8v g1;
    g1[0] = (int)(1u << 16);                                   // data_size=2B
    g1[1] = (int)(((unsigned)K_ & 0xffffu) << 16);             // tensor_dim0[15:0]
    g1[2] = (int)((((unsigned)K_ >> 16) & 0xffffu) | (1024u << 16)); // dim0 hi | tensor_dim1 lo (M)
    g1[3] = (int)(32u << 16);                                  // tensor_dim1 hi | tile_dim0=32
    g1[4] = (int)(128u);                                       // tile_dim1=128, tile_dim2=0
    g1[5] = (int)K_;                                           // tensor_dim0_stride lo32
    g1[6] = 0;                                                 // stride hi | dim1_stride lo
    g1[7] = 0;
    i32x4v z; z[0] = z[1] = z[2] = z[3] = 0;
#if defined(__clang_major__) && (__clang_major__ >= 23)
    i32x8v z8; for (int i = 0; i < 8; i++) z8[i] = 0;
    __builtin_amdgcn_tensor_load_to_lds(g0, g1, z, z, z8, 0);
#else
    __builtin_amdgcn_tensor_load_to_lds(g0, g1, z, z, 0);
#endif
}
#endif

// ---------- RMSNorm: fp32 in -> bf16 out ----------
__global__ __launch_bounds__(256)
void rmsnorm_kernel(const float* __restrict__ x, const float* __restrict__ w,
                    unsigned short* __restrict__ out) {
    __shared__ float red[256];
    const int t = blockIdx.x;
    const float* xr = x + (size_t)t * HDIM;
    float s = 0.f;
    for (int i = threadIdx.x; i < HDIM; i += 256) { float v = xr[i]; s += v * v; }
    red[threadIdx.x] = s;
    __syncthreads();
    for (int off = 128; off > 0; off >>= 1) {
        if (threadIdx.x < off) red[threadIdx.x] += red[threadIdx.x + off];
        __syncthreads();
    }
    const float r = rsqrtf(red[0] / (float)HDIM + 1e-6f);
    for (int i = threadIdx.x; i < HDIM; i += 256)
        out[(size_t)t * HDIM + i] = f2bf(w[i] * xr[i] * r);
}

// ---------- RoPE + layout [t][h*HD+d] -> [h][t][d], bf16; Q pre-scaled ----------
__global__ __launch_bounds__(128)
void rope_kernel(const float* __restrict__ q, const float* __restrict__ k,
                 const float* __restrict__ v,
                 unsigned short* __restrict__ Qb, unsigned short* __restrict__ Kb,
                 unsigned short* __restrict__ Vb) {
    const int t = blockIdx.x, h = blockIdx.y, d = threadIdx.x;
    const float qscale = 0.08838834764831845f; // 1/sqrt(128)
    const size_t src = (size_t)t * HDIM + h * HD + d;
    const size_t dst = ((size_t)h * T + t) * HD + d;
    const int dr = d & 63;
    const float inv = __powf(10000.f, -(2.f * (float)dr) / (float)HD);
    float sn, cs;
    __sincosf((float)t * inv, &sn, &cs);
    const float qa = q[src], ka = k[src];
    float qo, ko;
    if (d < 64) {
        const float qb = q[src + 64], kb = k[src + 64];
        qo = qa * cs - qb * sn;  ko = ka * cs - kb * sn;
    } else {
        const float qb = q[src - 64], kb = k[src - 64];
        qo = qa * cs + qb * sn;  ko = ka * cs + kb * sn;
    }
    Qb[dst] = f2bf(qo * qscale);
    Kb[dst] = f2bf(ko);
    Vb[dst] = f2bf(v[src]);
}

// ---------- Flash attention: 1 wave handles 16 query rows of one head ----------
__global__ __launch_bounds__(32)
void attn_kernel(const unsigned short* __restrict__ Qb,
                 const unsigned short* __restrict__ Kb,
                 const unsigned short* __restrict__ Vb,
                 unsigned short* __restrict__ attnb) {
    __shared__ __align__(16) unsigned short Ks[32 * HD];
    __shared__ __align__(16) unsigned short Vs[32 * HD];
    __shared__ __align__(16) unsigned short Ps[16 * 32];
    const int lane = threadIdx.x;
    const int q0 = blockIdx.x * 16;
    const int h  = blockIdx.y;
    const int half = lane >> 4, l15 = lane & 15;
    const unsigned short* Qh = Qb + (size_t)h * T * HD;
    const unsigned short* Kh = Kb + (size_t)h * T * HD;
    const unsigned short* Vh = Vb + (size_t)h * T * HD;

    U16 qf[4];
    for (int c = 0; c < 4; c++)
        for (int i = 0; i < 8; i++) {
            const int kk = 32 * c + (half << 4) + (i << 1);
            qf[c].u[i] = *(const unsigned*)&Qh[(size_t)(q0 + l15) * HD + kk];
        }

    v8f o[8];
    for (int dt = 0; dt < 8; dt++) for (int i = 0; i < 8; i++) o[dt][i] = 0.f;
    float mi[8], li[8];
    for (int i = 0; i < 8; i++) { mi[i] = -1e30f; li[i] = 0.f; }

    for (int key0 = 0; key0 <= q0 + 15; key0 += 32) {
        {
            const unsigned short* ks = Kh + (size_t)(key0 + lane) * HD;
            const unsigned short* vs = Vh + (size_t)(key0 + lane) * HD;
            for (int s4 = 0; s4 < 16; s4++) {
                ((uint4*)&Ks[lane * HD])[s4] = ((const uint4*)ks)[s4];
                ((uint4*)&Vs[lane * HD])[s4] = ((const uint4*)vs)[s4];
            }
        }
        __syncthreads();

        v8f s0, s1;
        for (int i = 0; i < 8; i++) { s0[i] = 0.f; s1[i] = 0.f; }
        for (int c = 0; c < 4; c++) {
            U16 b0, b1;
            for (int j = 0; j < 8; j++) {
                const int dd = 32 * c + (half << 4) + (j << 1);
                b0.u[j] = *(const unsigned*)&Ks[(size_t)l15 * HD + dd];
                b1.u[j] = *(const unsigned*)&Ks[(size_t)(16 + l15) * HD + dd];
            }
            s0 = __builtin_amdgcn_wmma_f32_16x16x32_bf16(false, qf[c].v, false, b0.v, (short)0, s0, false, false);
            s1 = __builtin_amdgcn_wmma_f32_16x16x32_bf16(false, qf[c].v, false, b1.v, (short)0, s1, false, false);
        }

        if (key0 + 31 > q0) {
            for (int i = 0; i < 8; i++) {
                const int row = q0 + i + 8 * half;
                if (key0 + l15 > row)      s0[i] = -1e30f;
                if (key0 + 16 + l15 > row) s1[i] = -1e30f;
            }
        }

        float sc[8];
        for (int i = 0; i < 8; i++) {
            float v = fmaxf(s0[i], s1[i]);
            for (int mk = 1; mk < 16; mk <<= 1) v = fmaxf(v, __shfl_xor(v, mk));
            const float mn = fmaxf(mi[i], v);
            sc[i] = __expf(mi[i] - mn);
            const float p0 = __expf(s0[i] - mn);
            const float p1 = __expf(s1[i] - mn);
            s0[i] = p0; s1[i] = p1;
            float rs = p0 + p1;
            for (int mk = 1; mk < 16; mk <<= 1) rs += __shfl_xor(rs, mk);
            li[i] = li[i] * sc[i] + rs;
            mi[i] = mn;
        }
        for (int dt = 0; dt < 8; dt++)
            for (int i = 0; i < 8; i++) o[dt][i] = o[dt][i] * sc[i];

        __syncthreads();
        for (int i = 0; i < 8; i++) {
            Ps[(i + 8 * half) * 32 + l15]      = f2bf(s0[i]);
            Ps[(i + 8 * half) * 32 + 16 + l15] = f2bf(s1[i]);
        }
        __syncthreads();
        U16 pf;
        for (int i = 0; i < 8; i++) {
            const int kk = (half << 4) + (i << 1);
            pf.u[i] = *(const unsigned*)&Ps[l15 * 32 + kk];
        }

        for (int dt = 0; dt < 8; dt++) {
            U16 vf;
            for (int j = 0; j < 8; j++) {
                const int kk = (half << 4) + (j << 1);
                const unsigned short a0 = Vs[(size_t)kk * HD + dt * 16 + l15];
                const unsigned short a1 = Vs[(size_t)(kk + 1) * HD + dt * 16 + l15];
                vf.u[j] = (unsigned)a0 | ((unsigned)a1 << 16);
            }
            o[dt] = __builtin_amdgcn_wmma_f32_16x16x32_bf16(false, pf.v, false, vf.v, (short)0, o[dt], false, false);
        }
        __syncthreads();
    }

    for (int dt = 0; dt < 8; dt++)
        for (int i = 0; i < 8; i++) {
            const int row = q0 + i + 8 * half;
            const int d = dt * 16 + l15;
            attnb[(size_t)row * HDIM + h * HD + d] = f2bf(o[dt][i] / li[i]);
        }
}

// ---------- Router: softmax over 64 logits + top-6 + normalized dense routing ----------
__global__ __launch_bounds__(32)
void router_kernel(const float* __restrict__ logits, float* __restrict__ routing) {
    const int t = blockIdx.x, lane = threadIdx.x;
    float p0 = logits[(size_t)t * NEXP + lane];
    float p1 = logits[(size_t)t * NEXP + 32 + lane];
    float m = fmaxf(p0, p1);
    for (int mk = 1; mk < 32; mk <<= 1) m = fmaxf(m, __shfl_xor(m, mk));
    p0 = __expf(p0 - m); p1 = __expf(p1 - m);
    float s = p0 + p1;
    for (int mk = 1; mk < 32; mk <<= 1) s += __shfl_xor(s, mk);
    p0 /= s; p1 /= s;
    float w0 = 0.f, w1 = 0.f, sumtop = 0.f;
    for (int it = 0; it < 6; it++) {
        float v = p0; int idx = lane;
        if (p1 > v) { v = p1; idx = lane + 32; }
        for (int mk = 1; mk < 32; mk <<= 1) {
            const float ov = __shfl_xor(v, mk);
            const int   oi = __shfl_xor(idx, mk);
            if (ov > v || (ov == v && oi < idx)) { v = ov; idx = oi; }
        }
        sumtop += v;
        if (idx == lane)           { w0 = p0; p0 = -1.f; }
        else if (idx == lane + 32) { w1 = p1; p1 = -1.f; }
    }
    routing[(size_t)t * NEXP + lane]      = w0 / sumtop;
    routing[(size_t)t * NEXP + 32 + lane] = w1 / sumtop;
}

// ---------- WMMA GEMM: C(MxN) = A_bf16(MxK) * B_fp32(KxN), fused epilogues ----------
// Block tile 128x64, 8 waves in 4x2 grid, wave tile 32x32 (4 WMMAs/wave/k-step).
// A tiles streamed by the Tensor Data Mover into double-buffered LDS.
// B tile stored in LDS as packed bf16 K-pairs -> all-32-bit LDS reads.
// bmode 0: B[k][n] = B[k*ldb+n]; bmode 2: B[k][n] = dpe[(k>>6)][n][k&63]
#define OP_STORE   0
#define OP_SILU    1
#define OP_MUL     2
#define OP_ADDBASE 3
#define OP_ACC     4

__global__ __launch_bounds__(256)
void wmma_gemm(const unsigned short* __restrict__ A, const float* __restrict__ B,
               int M, int N, int K, int ldb, int bmode, int op,
               float* __restrict__ Cf, unsigned short* __restrict__ Cb,
               const unsigned short* __restrict__ auxb,
               const float* __restrict__ route,
               const float* __restrict__ base) {
    __shared__ __align__(16) unsigned short Abuf[2][128 * 32]; // 2 x 8KB ping-pong
    __shared__ __align__(16) unsigned int   Bs2[16 * 64];      // packed K-pairs, 4KB
    const int tid = threadIdx.x;
    const int lane = tid & 31, wave = tid >> 5;
    const int wr = wave >> 1, wc = wave & 1;   // 4x2 waves; wave tile 32x32
    const int m0 = blockIdx.x * 128, n0 = blockIdx.y * 64;
    const int half = lane >> 4, l15 = lane & 15;

    v8f acc00, acc01, acc10, acc11;
    for (int i = 0; i < 8; i++) { acc00[i] = 0.f; acc01[i] = 0.f; acc10[i] = 0.f; acc11[i] = 0.f; }

#if USE_TDM
    if (wave == 0) tdm_load_A(A + (size_t)m0 * K, (unsigned)(uintptr_t)&Abuf[0][0], K);
    int buf = 0;
#endif

    for (int kb = 0; kb < K; kb += 32) {
        __syncthreads();
        // ---- stage B tile (32x64 fp32 -> packed bf16 pairs) + prefetch ----
        if (bmode == 0) {
            const int kp = tid >> 4;            // 0..15 K-pair
            const int c  = (tid & 15) * 4;      // 0..60 col
            const float* src0 = B + (size_t)(kb + 2 * kp) * ldb + n0 + c;
            const float4 f0 = *(const float4*)src0;          // row 2kp
            const float4 f1 = *(const float4*)(src0 + ldb);  // row 2kp+1
            if (kb + 32 < K) __builtin_prefetch(src0 + (size_t)32 * ldb, 0, 1);
            unsigned pk[4] = { pack2bf(f0.x, f1.x), pack2bf(f0.y, f1.y),
                               pack2bf(f0.z, f1.z), pack2bf(f0.w, f1.w) };
            *(uint4*)&Bs2[kp * 64 + c] = *(const uint4*)pk;
        } else {
            const int n = tid >> 2, ks = (tid & 3) * 8;     // 8 contiguous k for one n
            const long kg = (long)kb + ks;
            const float* src = B + (size_t)(kg >> 6) * DPE_STRIDE
                                 + (size_t)(n0 + n) * NEXP + (kg & 63);
            const float4 f0 = *(const float4*)src;
            const float4 f1 = *(const float4*)(src + 4);
            if (kb + 32 < K) {
                const long kn = kg + 32;
                __builtin_prefetch(B + (size_t)(kn >> 6) * DPE_STRIDE
                                     + (size_t)(n0 + n) * NEXP + (kn & 63), 0, 1);
            }
            const float fv[8] = { f0.x, f0.y, f0.z, f0.w, f1.x, f1.y, f1.z, f1.w };
            for (int j = 0; j < 4; j++)
                Bs2[((ks >> 1) + j) * 64 + n] = pack2bf(fv[2 * j], fv[2 * j + 1]);
        }
        // ---- A tile ready? (TDM path) or stage it with vector loads ----
#if USE_TDM
        if (wave == 0) __builtin_amdgcn_s_wait_tensorcnt(0);
#else
        {
            const int r = tid >> 1, c = (tid & 1) * 16;
            const unsigned short* srcA = A + (size_t)(m0 + r) * K + kb + c;
            *(uint4*)&Abuf[0][r * 32 + c]     = *(const uint4*)srcA;
            *(uint4*)&Abuf[0][r * 32 + c + 8] = *(const uint4*)(srcA + 8);
        }
#endif
        __syncthreads();
#if USE_TDM
        if (wave == 0 && kb + 32 < K)
            tdm_load_A(A + (size_t)m0 * K + kb + 32,
                       (unsigned)(uintptr_t)&Abuf[buf ^ 1][0], K);
        const unsigned short* Acur = Abuf[buf];
        buf ^= 1;
#else
        const unsigned short* Acur = Abuf[0];
#endif

        // ---- fragments (all 32-bit LDS reads) ----
        U16 a0, a1, b0, b1;
        for (int i = 0; i < 8; i++) {
            const int kk = (half << 4) + (i << 1);
            a0.u[i] = *(const unsigned*)&Acur[(wr * 32 + l15) * 32 + kk];
            a1.u[i] = *(const unsigned*)&Acur[(wr * 32 + 16 + l15) * 32 + kk];
        }
        for (int j = 0; j < 8; j++) {
            const int kp = half * 8 + j;        // packed pair index: K = 2*kp, 2*kp+1
            b0.u[j] = Bs2[kp * 64 + wc * 32 + l15];
            b1.u[j] = Bs2[kp * 64 + wc * 32 + 16 + l15];
        }
        acc00 = __builtin_amdgcn_wmma_f32_16x16x32_bf16(false, a0.v, false, b0.v, (short)0, acc00, false, false);
        acc01 = __builtin_amdgcn_wmma_f32_16x16x32_bf16(false, a0.v, false, b1.v, (short)0, acc01, false, false);
        acc10 = __builtin_amdgcn_wmma_f32_16x16x32_bf16(false, a1.v, false, b0.v, (short)0, acc10, false, false);
        acc11 = __builtin_amdgcn_wmma_f32_16x16x32_bf16(false, a1.v, false, b1.v, (short)0, acc11, false, false);
    }

    // ---- fused epilogue ----
    for (int ai = 0; ai < 2; ai++) {
        for (int g = 0; g < 2; g++) {
            for (int i = 0; i < 8; i++) {
                const int row = m0 + wr * 32 + ai * 16 + half * 8 + i;
                const int col = n0 + wc * 32 + g * 16 + l15;
                float v;
                if (ai == 0) v = (g == 0) ? acc00[i] : acc01[i];
                else         v = (g == 0) ? acc10[i] : acc11[i];
                const size_t idx = (size_t)row * N + col;
                switch (op) {
                    case OP_STORE:
                        if (Cf) Cf[idx] = v;
                        if (Cb) Cb[idx] = f2bf(v);
                        break;
                    case OP_SILU: {
                        const float sg = v / (1.f + __expf(-v));
                        Cb[idx] = f2bf(sg);
                    } break;
                    case OP_MUL: {
                        float w = v * bf2f(auxb[idx]);
                        if (route) w *= route[(size_t)row * NEXP + (col & 63)];
                        Cb[idx] = f2bf(w);
                    } break;
                    case OP_ADDBASE:
                        Cf[idx] = v + base[idx];
                        break;
                    case OP_ACC:
                        Cf[idx] += v;
                        break;
                }
            }
        }
    }
}

// ---------------------- host launcher ----------------------
extern "C" void kernel_launch(void* const* d_in, const int* in_sizes, int n_in,
                              void* d_out, int out_size, void* d_ws, size_t ws_size,
                              hipStream_t stream) {
    (void)in_sizes; (void)n_in; (void)out_size; (void)ws_size;
    const float* x       = (const float*)d_in[0];
    const float* ln1_w   = (const float*)d_in[1];
    const float* ln2_w   = (const float*)d_in[2];
    const float* Wq      = (const float*)d_in[3];
    const float* Wk      = (const float*)d_in[4];
    const float* Wv      = (const float*)d_in[5];
    const float* Wo      = (const float*)d_in[6];
    const float* gate_w  = (const float*)d_in[7];
    const float* gpe     = (const float*)d_in[8];
    const float* upe     = (const float*)d_in[9];
    const float* dpe     = (const float*)d_in[10];
    const float* sh_gate = (const float*)d_in[11];
    const float* sh_up   = (const float*)d_in[12];
    const float* sh_down = (const float*)d_in[13];
    float* out = (float*)d_out;

    char* p = (char*)d_ws;
    auto alloc = [&](size_t bytes) -> void* {
        void* r = (void*)p;
        p += (bytes + 255) & ~(size_t)255;
        return r;
    };
    unsigned short* xn1   = (unsigned short*)alloc((size_t)T * HDIM * 2);
    float* qf    = (float*)alloc((size_t)T * HDIM * 4);
    float* kf    = (float*)alloc((size_t)T * HDIM * 4);
    float* vf    = (float*)alloc((size_t)T * HDIM * 4);
    unsigned short* Qb    = (unsigned short*)alloc((size_t)NHEAD * T * HD * 2);
    unsigned short* Kb    = (unsigned short*)alloc((size_t)NHEAD * T * HD * 2);
    unsigned short* Vb    = (unsigned short*)alloc((size_t)NHEAD * T * HD * 2);
    unsigned short* attnb = (unsigned short*)alloc((size_t)T * HDIM * 2);
    float* hbuf  = (float*)alloc((size_t)T * HDIM * 4);
    unsigned short* xn2   = (unsigned short*)alloc((size_t)T * HDIM * 2);
    float* logits  = (float*)alloc((size_t)T * NEXP * 4);
    float* routing = (float*)alloc((size_t)T * NEXP * 4);
    unsigned short* Gbuf  = (unsigned short*)alloc((size_t)T * GU_N * 2); // 117 MB
    unsigned short* Sg    = (unsigned short*)alloc((size_t)T * SHF * 2);

    // ---- attention block ----
    rmsnorm_kernel<<<T, 256, 0, stream>>>(x, ln1_w, xn1);
    wmma_gemm<<<dim3(T/128, HDIM/64), 256, 0, stream>>>(xn1, Wq, T, HDIM, HDIM, HDIM, 0,
        OP_STORE, qf, nullptr, nullptr, nullptr, nullptr);
    wmma_gemm<<<dim3(T/128, HDIM/64), 256, 0, stream>>>(xn1, Wk, T, HDIM, HDIM, HDIM, 0,
        OP_STORE, kf, nullptr, nullptr, nullptr, nullptr);
    wmma_gemm<<<dim3(T/128, HDIM/64), 256, 0, stream>>>(xn1, Wv, T, HDIM, HDIM, HDIM, 0,
        OP_STORE, vf, nullptr, nullptr, nullptr, nullptr);
    rope_kernel<<<dim3(T, NHEAD), 128, 0, stream>>>(qf, kf, vf, Qb, Kb, Vb);
    attn_kernel<<<dim3(T/16, NHEAD), 32, 0, stream>>>(Qb, Kb, Vb, attnb);
    wmma_gemm<<<dim3(T/128, HDIM/64), 256, 0, stream>>>(attnb, Wo, T, HDIM, HDIM, HDIM, 0,
        OP_ADDBASE, hbuf, nullptr, nullptr, nullptr, x);   // h = x + attn

    // ---- MoE block ----
    rmsnorm_kernel<<<T, 256, 0, stream>>>(hbuf, ln2_w, xn2);
    wmma_gemm<<<dim3(T/128, NEXP/64), 256, 0, stream>>>(xn2, gate_w, T, NEXP, HDIM, NEXP, 0,
        OP_STORE, logits, nullptr, nullptr, nullptr, nullptr);
    router_kernel<<<T, 32, 0, stream>>>(logits, routing);

    // dense all-expert gate/up (coalesced n = i*64+e layout), fused silu & gate*up*routing
    wmma_gemm<<<dim3(T/128, GU_N/64), 256, 0, stream>>>(xn2, gpe, T, GU_N, HDIM, GU_N, 0,
        OP_SILU, nullptr, Gbuf, nullptr, nullptr, nullptr);
    wmma_gemm<<<dim3(T/128, GU_N/64), 256, 0, stream>>>(xn2, upe, T, GU_N, HDIM, GU_N, 0,
        OP_MUL, nullptr, Gbuf, Gbuf, routing, nullptr);    // in-place: W = silu(g)*u*route

    // shared FFN gate/up
    wmma_gemm<<<dim3(T/128, SHF/64), 256, 0, stream>>>(xn2, sh_gate, T, SHF, HDIM, SHF, 0,
        OP_SILU, nullptr, Sg, nullptr, nullptr, nullptr);
    wmma_gemm<<<dim3(T/128, SHF/64), 256, 0, stream>>>(xn2, sh_up, T, SHF, HDIM, SHF, 0,
        OP_MUL, nullptr, Sg, Sg, nullptr, nullptr);

    // routed down-proj (dpe block layout, bmode 2): out = h + W @ Dp
    wmma_gemm<<<dim3(T/128, HDIM/64), 256, 0, stream>>>(Gbuf, dpe, T, HDIM, GU_N, 0, 2,
        OP_ADDBASE, out, nullptr, nullptr, nullptr, hbuf);
    // shared down-proj accumulate: out += Sg @ sh_down
    wmma_gemm<<<dim3(T/128, HDIM/64), 256, 0, stream>>>(Sg, sh_down, T, HDIM, SHF, HDIM, 0,
        OP_ACC, out, nullptr, nullptr, nullptr, nullptr);
}
